// CausalSelfAttention_34093450395829
// MI455X (gfx1250) — matrix operand order repaired and checked by reading, not compiled, not run on hardware
//
#include <hip/hip_runtime.h>
#include <hip/hip_bf16.h>
#include <cstdint>

// ---------------------------------------------------------------------------
// CDNA5 (gfx1250) causal self-attention, bf16 WMMA, f32 accumulation.
// B=4, T=2048, C=1024, H=16, hd=64.
//   GEMM tiles: 128x64 per block (8 waves, 4 wmma/wave/K-step, A-frag reuse x4)
//   Attention: 64-key streaming blocks, online softmax, 16 wmma per block/wave
//   Staging: GLOBAL_LOAD_ASYNC_TO_LDS_B128 when the builtin exists.
// ---------------------------------------------------------------------------

typedef __attribute__((ext_vector_type(16))) __bf16 v16bf;
typedef __attribute__((ext_vector_type(8)))  float  v8f;
typedef __attribute__((ext_vector_type(4)))  int    v4i;

#define CB  1024
#define CT  2048
#define CH  16
#define CHD 64

// ---- async global->LDS staging (gfx1250 ASYNCcnt path), guarded ----------
#if defined(__AMDGCN__) && \
    __has_builtin(__builtin_amdgcn_global_load_async_to_lds_b128) && \
    __has_builtin(__builtin_amdgcn_s_wait_asynccnt)
#define USE_ASYNC_LDS 1
#endif

typedef __attribute__((address_space(1))) v4i glob_v4i;   // global int4
typedef __attribute__((address_space(3))) v4i lds_v4i;    // LDS int4

__device__ inline void async_copy_b128(void* lds_dst, const void* gsrc) {
#if defined(USE_ASYNC_LDS)
    __builtin_amdgcn_global_load_async_to_lds_b128(
        (glob_v4i*)(uintptr_t)gsrc, (lds_v4i*)(uintptr_t)lds_dst, 0, 0);
#else
    *(uint4*)lds_dst = *(const uint4*)gsrc;
#endif
}

__device__ inline void async_copy_wait() {
#if defined(USE_ASYNC_LDS)
    __builtin_amdgcn_s_wait_asynccnt(0);
#endif
}

// ---- fragment helpers -----------------------------------------------------
union Frag2x8 { v16bf v; uint4 q[2]; };

__device__ inline v16bf load_frag2(const __bf16* p0, const __bf16* p1) {
    Frag2x8 u;
    u.q[0] = *(const uint4*)p0;
    u.q[1] = *(const uint4*)p1;
    return u.v;
}

__device__ inline v8f wmma_bf16(v16bf a, v16bf b, v8f c) {
    return __builtin_amdgcn_wmma_f32_16x16x32_bf16(
        false, a, false, b, (short)0, c, false, false);
}

// ---------------------------------------------------------------------------
// f32 -> bf16 conversion (grid-stride)
// ---------------------------------------------------------------------------
__global__ void f32_to_bf16_kernel(const float* __restrict__ in,
                                   __bf16* __restrict__ out, int n) {
    int i = blockIdx.x * blockDim.x + threadIdx.x;
    int stride = gridDim.x * blockDim.x;
    for (; i < n; i += stride) out[i] = (__bf16)in[i];
}

// ---------------------------------------------------------------------------
// GEMM: C[M,N] = A[M,K]*B[K,N] + bias.  128x64 block tile, 256 thr = 8 waves.
// Wave w owns rows [m0+16w, m0+16w+16), all 4 column sub-tiles (4 wmma/step).
// MODE 0: scatter bf16 q/k/v into [B,H,T,hd];  MODE 1: f32 out.
// ---------------------------------------------------------------------------
__device__ inline void scatter_qkv(__bf16* __restrict__ q, __bf16* __restrict__ k,
                                   __bf16* __restrict__ v, int mm, int nn, float val) {
    int s  = nn >> 10;
    int cc = nn & (CB - 1);
    int h  = cc >> 6;
    int d  = cc & (CHD - 1);
    int b  = mm >> 11;
    int t  = mm & (CT - 1);
    size_t idx = ((((size_t)b * CH + h) * CT) + t) * CHD + d;
    __bf16 bv = (__bf16)val;
    if (s == 0) q[idx] = bv;
    else if (s == 1) k[idx] = bv;
    else v[idx] = bv;
}

template <int MODE>
__global__ __launch_bounds__(256) void gemm128x64_kernel(
    const __bf16* __restrict__ A, const __bf16* __restrict__ Bm,
    const float* __restrict__ bias, float* __restrict__ outF,
    __bf16* __restrict__ outQ, __bf16* __restrict__ outK,
    __bf16* __restrict__ outV, int M, int N, int K) {

    constexpr int LDA = 40;   // 128x32 A tile (80B rows, 16B aligned)
    constexpr int LDB = 40;   // 64x32 B tile, stored transposed [n][k]
    __shared__ __bf16 sA[128 * LDA];
    __shared__ __bf16 sB[64 * LDB];

    const int tid    = threadIdx.x;
    const int wave   = tid >> 5;
    const int lane   = tid & 31;
    const int laneLo = lane & 15;
    const int hi     = lane >> 4;
    const int m0     = blockIdx.y * 128;
    const int n0     = blockIdx.x * 64;

    v8f acc[4] = {};

    for (int k0 = 0; k0 < K; k0 += 32) {
        __syncthreads();
        // ---- stage A tile 128x32: 2 async b128 per thread
        {
            int r = tid >> 1, cb = (tid & 1) * 16;
            const __bf16* src = A + (size_t)(m0 + r) * K + k0 + cb;
            __bf16* dst = sA + r * LDA + cb;
            async_copy_b128(dst, src);
            async_copy_b128(dst + 8, src + 8);
        }
        // ---- stage B tile 32x64 transposed -> sB[n][k]
        {
            int idx = tid * 8;
            int r = idx >> 6;   // k row
            int c = idx & 63;   // n col
            union { uint4 q; __bf16 h[8]; } u;
            u.q = *(const uint4*)(Bm + (size_t)(k0 + r) * N + n0 + c);
#pragma unroll
            for (int i = 0; i < 8; ++i) sB[(c + i) * LDB + r] = u.h[i];
        }
        if (k0 + 32 < K) {
            __builtin_prefetch(A + (size_t)(m0 + (tid & 127)) * K + k0 + 32, 0, 1);
            __builtin_prefetch(Bm + (size_t)(k0 + 32 + (tid & 31)) * N + n0, 0, 1);
        }
        async_copy_wait();
        __syncthreads();

        const __bf16* aRow = sA + (wave * 16 + laneLo) * LDA;
        v16bf af = load_frag2(aRow + hi * 8, aRow + 16 + hi * 8);
#pragma unroll
        for (int n = 0; n < 4; ++n) {
            const __bf16* bRow = sB + (n * 16 + laneLo) * LDB;
            acc[n] = wmma_bf16(af, load_frag2(bRow + hi * 8, bRow + 16 + hi * 8), acc[n]);
        }
    }

    // ---- epilogue
#pragma unroll
    for (int n = 0; n < 4; ++n) {
        const int nn = n0 + n * 16 + laneLo;
        const float bv = bias[nn];
#pragma unroll
        for (int r = 0; r < 8; ++r) {
            int mm = m0 + wave * 16 + hi * 8 + r;
            float val = acc[n][r] + bv;
            if (MODE == 1) outF[(size_t)mm * N + nn] = val;
            else           scatter_qkv(outQ, outK, outV, mm, nn, val);
        }
    }
}

// ---------------------------------------------------------------------------
// Flash attention (causal). Block = (b,h) x 64 q-rows; 128 thr = 4 waves.
// Wave w owns q rows [q0+16w, q0+16w+16). Keys streamed in 64-wide blocks:
//   S(16x64) = Q*K^T   : 4 col-tiles x 2 K-chunks = 8 wmma
//   online softmax     : one shuffle-reduction pass per 8 rows per block
//   O += P(16x64)*V    : 4 ch-tiles x 2 key-chunks = 8 wmma
// All key blocks (kb <= q0) intersect every wave's rows -> no divergent skip.
// ---------------------------------------------------------------------------
__global__ __launch_bounds__(128) void flash_attn_kernel(
    const __bf16* __restrict__ Qm, const __bf16* __restrict__ Km,
    const __bf16* __restrict__ Vm, __bf16* __restrict__ Y) {

    constexpr int LDQ = 72;   // 144B rows
    constexpr int LDK = 64;   // [key][ch], 128B rows
    constexpr int LDV = 72;   // V transposed [ch][key], 144B rows
    __shared__ __bf16 sQ[64 * LDQ];
    __shared__ __bf16 sK[64 * LDK];
    __shared__ __bf16 sVt[64 * LDV];
    __shared__ __bf16 sP[4][16 * 64];

    const int bh     = blockIdx.y;             // b*H + h
    const int q0     = blockIdx.x * 64;
    const int tid    = threadIdx.x;
    const int wave   = tid >> 5;
    const int lane   = tid & 31;
    const int laneLo = lane & 15;
    const int hi     = lane >> 4;
    const size_t base = (size_t)bh * CT * CHD;

    // ---- stage Q tile 64x64 (async)
    {
        int row = tid >> 1, cb = (tid & 1) * 32;
        const __bf16* src = Qm + base + (size_t)(q0 + row) * CHD + cb;
        __bf16* dst = sQ + row * LDQ + cb;
#pragma unroll
        for (int i = 0; i < 4; ++i) async_copy_b128(dst + i * 8, src + i * 8);
    }
    async_copy_wait();
    __syncthreads();

    const __bf16* qRow = sQ + (wave * 16 + laneLo) * LDQ;
    v16bf qf0 = load_frag2(qRow + hi * 8,      qRow + 16 + hi * 8);  // ch 0..31
    v16bf qf1 = load_frag2(qRow + 32 + hi * 8, qRow + 48 + hi * 8);  // ch 32..63

    v8f o[4] = {};
    float rmax[8], rsum[8];
#pragma unroll
    for (int r = 0; r < 8; ++r) { rmax[r] = -3.0e38f; rsum[r] = 0.0f; }

    const int   qrow_lo = q0 + wave * 16;
    const int   kend    = q0 + 64;
    const float scale   = 0.125f;   // 1/sqrt(64)

    for (int kb = 0; kb < kend; kb += 64) {
        __syncthreads();
        // ---- stage K block [64][64] row-major (async)
        {
            int row = tid >> 1, cb = (tid & 1) * 32;
            const __bf16* src = Km + base + (size_t)(kb + row) * CHD + cb;
            __bf16* dst = sK + row * LDK + cb;
#pragma unroll
            for (int i = 0; i < 4; ++i) async_copy_b128(dst + i * 8, src + i * 8);
        }
        // ---- stage V block transposed [ch][key]
        {
            int r = tid >> 1, cb = (tid & 1) * 32;
            union { uint4 q[4]; __bf16 h[32]; } u;
            const uint4* s = (const uint4*)(Vm + base + (size_t)(kb + r) * CHD + cb);
            u.q[0] = s[0]; u.q[1] = s[1]; u.q[2] = s[2]; u.q[3] = s[3];
#pragma unroll
            for (int i = 0; i < 32; ++i) sVt[(cb + i) * LDV + r] = u.h[i];
        }
        async_copy_wait();
        __syncthreads();

        // ---- S = Q * K^T for 64 keys
        v8f sc[4];
#pragma unroll
        for (int t = 0; t < 4; ++t) {
            const __bf16* kR = sK + (t * 16 + laneLo) * LDK;
            v16bf k0 = load_frag2(kR + hi * 8,      kR + 16 + hi * 8);
            v16bf k1 = load_frag2(kR + 32 + hi * 8, kR + 48 + hi * 8);
            v8f s = {};
            s = wmma_bf16(qf0, k0, s);
            s = wmma_bf16(qf1, k1, s);
            sc[t] = s;
        }

        // ---- causal mask + online softmax (row m = hi*8 + r, col = t*16+laneLo)
#pragma unroll
        for (int r = 0; r < 8; ++r) {
            const int qi = qrow_lo + hi * 8 + r;
            float vv[4];
#pragma unroll
            for (int t = 0; t < 4; ++t) {
                int ki = kb + t * 16 + laneLo;
                vv[t] = (ki <= qi) ? sc[t][r] * scale : -3.0e38f;
            }
            float mnew = rmax[r];
#pragma unroll
            for (int t = 0; t < 4; ++t) mnew = fmaxf(mnew, vv[t]);
#pragma unroll
            for (int mask = 1; mask <= 8; mask <<= 1)
                mnew = fmaxf(mnew, __shfl_xor(mnew, mask, 32));

            float psum = 0.0f;
#pragma unroll
            for (int t = 0; t < 4; ++t) {
                float p = __expf(vv[t] - mnew);
                psum += p;
                sP[wave][(hi * 8 + r) * 64 + t * 16 + laneLo] = (__bf16)p;
            }
#pragma unroll
            for (int mask = 1; mask <= 8; mask <<= 1)
                psum += __shfl_xor(psum, mask, 32);

            float corr = __expf(rmax[r] - mnew);
            rmax[r] = mnew;
            rsum[r] = rsum[r] * corr + psum;
            o[0][r] *= corr; o[1][r] *= corr; o[2][r] *= corr; o[3][r] *= corr;
        }

        // ---- O += P(16x64) * V(64x64)
        const __bf16* pR = &sP[wave][laneLo * 64];
        v16bf pf0 = load_frag2(pR + hi * 8,      pR + 16 + hi * 8);   // keys 0..31
        v16bf pf1 = load_frag2(pR + 32 + hi * 8, pR + 48 + hi * 8);   // keys 32..63
#pragma unroll
        for (int t = 0; t < 4; ++t) {
            const __bf16* vR = sVt + (t * 16 + laneLo) * LDV;
            v16bf v0 = load_frag2(vR + hi * 8,      vR + 16 + hi * 8);
            v16bf v1 = load_frag2(vR + 32 + hi * 8, vR + 48 + hi * 8);
            o[t] = wmma_bf16(pf0, v0, o[t]);
            o[t] = wmma_bf16(pf1, v1, o[t]);
        }
    }

    // ---- normalize + store Y as [B,T,C] bf16
    const int b = bh >> 4;
    const int h = bh & 15;
#pragma unroll
    for (int r = 0; r < 8; ++r) {
        const int   qi  = qrow_lo + hi * 8 + r;
        const float inv = 1.0f / rsum[r];
        __bf16* yRow = Y + ((size_t)b * CT + qi) * CB + h * CHD;
#pragma unroll
        for (int t = 0; t < 4; ++t)
            yRow[t * 16 + laneLo] = (__bf16)(o[t][r] * inv);
    }
}

// ---------------------------------------------------------------------------
// Launch
// ---------------------------------------------------------------------------
extern "C" void kernel_launch(void* const* d_in, const int* in_sizes, int n_in,
                              void* d_out, int out_size, void* d_ws, size_t ws_size,
                              hipStream_t stream) {
    (void)in_sizes; (void)n_in; (void)out_size; (void)ws_size;

    const int T = CT, C = CB;
    const int M = 4 * T;          // 8192
    const int N3 = 3 * C;         // 3072

    const float* x      = (const float*)d_in[0];
    const float* W_attn = (const float*)d_in[1];
    const float* b_attn = (const float*)d_in[2];
    const float* W_proj = (const float*)d_in[3];
    const float* b_proj = (const float*)d_in[4];

    char* ws = (char*)d_ws;
    size_t off = 0;
    auto carve = [&](size_t bytes) {
        char* p = ws + off;
        off = (off + bytes + 255) & ~(size_t)255;
        return p;
    };
    __bf16* xb  = (__bf16*)carve((size_t)M * C * 2);
    __bf16* wab = (__bf16*)carve((size_t)C * N3 * 2);
    __bf16* wpb = (__bf16*)carve((size_t)C * C * 2);
    __bf16* qb  = (__bf16*)carve((size_t)M * C * 2);
    __bf16* kb  = (__bf16*)carve((size_t)M * C * 2);
    __bf16* vb  = (__bf16*)carve((size_t)M * C * 2);
    __bf16* yb  = (__bf16*)carve((size_t)M * C * 2);

    // 1) f32 -> bf16
    f32_to_bf16_kernel<<<2048, 256, 0, stream>>>(x, xb, M * C);
    f32_to_bf16_kernel<<<1024, 256, 0, stream>>>(W_attn, wab, C * N3);
    f32_to_bf16_kernel<<<512, 256, 0, stream>>>(W_proj, wpb, C * C);

    // 2) QKV GEMM -> scatter q/k/v [B,H,T,hd] bf16
    gemm128x64_kernel<0><<<dim3(N3 / 64, M / 128), 256, 0, stream>>>(
        xb, wab, b_attn, nullptr, qb, kb, vb, M, N3, C);

    // 3) Flash attention
    flash_attn_kernel<<<dim3(T / 64, 4 * CH), 128, 0, stream>>>(qb, kb, vb, yb);

    // 4) Proj GEMM -> f32 out
    gemm128x64_kernel<1><<<dim3(C / 64, M / 128), 256, 0, stream>>>(
        yb, wpb, b_proj, (float*)d_out, nullptr, nullptr, nullptr, M, C, C);
}
